// RNNPolicy_17781164605687
// MI455X (gfx1250) — matrix operand order repaired and checked by reading, not compiled, not run on hardware
//
#include <hip/hip_runtime.h>

// ---------------------------------------------------------------------------
// MI455X (gfx1250) fused 2-layer RNN + MLP head.
//
// Roofline: ~22 GFLOP total, 134 MB input -> both trivial; the kernel is
// latency-bound on the T=2048 sequential scan. Strategy: 16 persistent
// workgroups (one per 16-row batch tile), 8 waves each (wave32):
//   waves 0-3: layer-0 recurrence (one 16x16 N-tile per wave, 4 WMMAs/step)
//   waves 4-7: layer-1 recurrence, pipelined one step behind layer 0
// Hidden state lives in LDS (f16, double buffered); weights live in VGPRs
// (v16h B-tiles) for the entire loop. One workgroup barrier per step.
// h1 is written to d_ws in WMMA-A-register byte layout (f16, 64 MiB) so the
// head kernel reads it back with raw b128 loads and feeds WMMA directly.
// ---------------------------------------------------------------------------

typedef _Float16 half_t;
typedef __attribute__((ext_vector_type(16))) _Float16 v16h;
typedef __attribute__((ext_vector_type(8)))  _Float16 v8h;
typedef __attribute__((ext_vector_type(8)))  float    v8f;
typedef __attribute__((ext_vector_type(4)))  float    v4f;

#define B_   256
#define T_   2048
#define I_   64
#define H_   64
#define O_   2
#define BT_TILES (B_ / 16)   // 16 batch tiles

// ---------------------------------------------------------------- helpers --

static __device__ __forceinline__ v8f wmma_f16(v16h a, v16h b, v8f c) {
  // D = A(16x32 f16) * B(32x16 f16) + C(16x16 f32)
  return __builtin_amdgcn_wmma_f32_16x16x32_f16(
      /*neg_a=*/false, a, /*neg_b=*/false, b,
      /*c_mod=*/(short)0, c, /*reuse_a=*/false, /*reuse_b=*/false);
}

static __device__ __forceinline__ v8f splat8(float v) {
  v8f r;
#pragma unroll
  for (int i = 0; i < 8; ++i) r[i] = v;
  return r;
}

static __device__ __forceinline__ v16h zero16h() {
  v16h z;
#pragma unroll
  for (int i = 0; i < 16; ++i) z[i] = (_Float16)0.0f;
  return z;
}

static __device__ __forceinline__ float tanh_fast(float v) {
#if __has_builtin(__builtin_amdgcn_tanhf)
  return __builtin_amdgcn_tanhf(v);        // v_tanh_f32 on gfx1250
#else
  float e = __expf(2.0f * v);
  return 1.0f - 2.0f / (e + 1.0f);
#endif
}

static __device__ __forceinline__ v16h cvt16(v4f a, v4f b, v4f c, v4f d) {
  v16h r;
#pragma unroll
  for (int i = 0; i < 4; ++i) {
    r[i]      = (_Float16)a[i];
    r[4 + i]  = (_Float16)b[i];
    r[8 + i]  = (_Float16)c[i];
    r[12 + i] = (_Float16)d[i];
  }
  return r;
}

// B-tile (32x16, K-major): lane p -> column n = nt*16 + (p&15);
// lanes 0-15 hold K = 32c+[0..7], 32c+[16..23]; lanes 16-31 hold K = 32c+[8..15], 32c+[24..31].
// W is f32 row-major (N=64, K=64): B[k][n] = W[n][k] -> contiguous reads.
static __device__ __forceinline__ v16h load_B_f32(const float* __restrict__ W,
                                                  int nt, int kc, int lane) {
  int n  = nt * 16 + (lane & 15);
  int k0 = kc * 32 + ((lane & 16) ? 8 : 0);
  const v4f* p = (const v4f*)(W + n * 64 + k0);
  return cvt16(p[0], p[1], p[4], p[5]);
}

// A-tile (16x32) from f32 row-major src with row stride rs (floats).
// lane<16: row=lane, K = 32c+[0..7], 32c+[16..23]
// lane>=16: row=lane-16, K = 32c+[8..15], 32c+[24..31]
static __device__ __forceinline__ v16h load_A_f32(const float* __restrict__ src,
                                                  long rs, int kc, int lane) {
  int m  = lane & 15;
  int k0 = kc * 32 + ((lane & 16) ? 8 : 0);
  const float* p = src + (long)m * rs + k0;
  v4f a0 = *(const v4f*)(p);
  v4f a1 = *(const v4f*)(p + 4);
  v4f a2 = *(const v4f*)(p + 16);
  v4f a3 = *(const v4f*)(p + 20);
  return cvt16(a0, a1, a2, a3);
}

// A-tile from an LDS-resident 16x64 f16 row-major tile (two ds_load_b128).
static __device__ __forceinline__ v16h load_A_lds(const half_t* tile,
                                                  int kc, int lane) {
  int m  = lane & 15;
  int k0 = kc * 32 + ((lane & 16) ? 8 : 0);
  const half_t* p = tile + m * 64 + k0;
  v8h lo = *(const v8h*)(p);
  v8h hi = *(const v8h*)(p + 16);
  v16h a;
#pragma unroll
  for (int i = 0; i < 8; ++i) { a[i] = lo[i]; a[8 + i] = hi[i]; }
  return a;
}

// Scatter a D-tile (f32 C/D layout) into a row-major f16 16x64 LDS tile.
static __device__ __forceinline__ void store_D_lds(half_t* tile, int nt,
                                                   int lane, v8f d) {
  int n  = nt * 16 + (lane & 15);
  int m0 = (lane & 16) ? 8 : 0;
#pragma unroll
  for (int i = 0; i < 8; ++i) tile[(m0 + i) * 64 + n] = (half_t)d[i];
}

// Persist / reload an A-register chunk in raw per-lane byte layout.
static __device__ __forceinline__ void store_A_ws(half_t* rec, int kc,
                                                  int lane, v16h a) {
  v8h lo, hi;
#pragma unroll
  for (int i = 0; i < 8; ++i) { lo[i] = a[i]; hi[i] = a[8 + i]; }
  v8h* q = (v8h*)(rec + (size_t)kc * 512 + lane * 16);
  q[0] = lo;
  q[1] = hi;
}

static __device__ __forceinline__ v16h load_A_ws(const half_t* rec, int kc,
                                                 int lane) {
  const v8h* q = (const v8h*)(rec + (size_t)kc * 512 + lane * 16);
  v8h lo = q[0], hi = q[1];
  v16h a;
#pragma unroll
  for (int i = 0; i < 8; ++i) { a[i] = lo[i]; a[8 + i] = hi[i]; }
  return a;
}

// ------------------------------------------------------- fused RNN kernel --

__global__ __launch_bounds__(256, 1)
void rnn2_scan_kernel(const float* __restrict__ x,
                      const float* __restrict__ Wih0, const float* __restrict__ Whh0,
                      const float* __restrict__ bih0, const float* __restrict__ bhh0,
                      const float* __restrict__ Wih1, const float* __restrict__ Whh1,
                      const float* __restrict__ bih1, const float* __restrict__ bhh1,
                      half_t* __restrict__ h1ws) {
  __shared__ __align__(16) half_t lds_h0[2][16 * 64];
  __shared__ __align__(16) half_t lds_h1[2][16 * 64];

  const int lane  = threadIdx.x & 31;
  const int w     = threadIdx.x >> 5;   // wave 0..7
  const int nt    = w & 3;              // N-tile of this wave
  const bool L0   = (w < 4);
  const int btile = blockIdx.x;         // 0..15
  const int b0    = btile * 16;
  const long xrs  = (long)T_ * I_;      // x row stride (floats)

  // Per-wave resident weights (4 x v16h = 32 VGPRs) + fused bias.
  const float* Wih = L0 ? Wih0 : Wih1;
  const float* Whh = L0 ? Whh0 : Whh1;
  v16h Bih0 = load_B_f32(Wih, nt, 0, lane);
  v16h Bih1 = load_B_f32(Wih, nt, 1, lane);
  v16h Bhh0 = load_B_f32(Whh, nt, 0, lane);
  v16h Bhh1 = load_B_f32(Whh, nt, 1, lane);
  int  nn   = nt * 16 + (lane & 15);
  v8f  cbv  = splat8(L0 ? (bih0[nn] + bhh0[nn]) : (bih1[nn] + bhh1[nn]));

  v16h Ahp0 = zero16h();   // h(t-1) of this wave's layer, A layout
  v16h Ahp1 = zero16h();

  for (int s = 0; s <= T_; ++s) {
    if (L0) {
      if (s < T_) {
        if (w == 0 && s + 24 < T_) {   // pull future x rows toward L2/WGP$
          const float* pf = x + ((long)(b0 + (lane & 15)) * T_ + (s + 24)) * I_ +
                            ((lane & 16) ? 32 : 0);
          __builtin_prefetch(pf, 0, 1);   // global_prefetch_b8
        }
        const float* src = x + (long)b0 * xrs + (long)s * I_;
        v16h Ax0 = load_A_f32(src, xrs, 0, lane);
        v16h Ax1 = load_A_f32(src, xrs, 1, lane);
        v8f acc = cbv;
        acc = wmma_f16(Ax0, Bih0, acc);
        acc = wmma_f16(Ax1, Bih1, acc);
        acc = wmma_f16(Ahp0, Bhh0, acc);
        acc = wmma_f16(Ahp1, Bhh1, acc);
        v8f th;
#pragma unroll
        for (int i = 0; i < 8; ++i) th[i] = tanh_fast(acc[i]);
        store_D_lds(lds_h0[s & 1], nt, lane, th);
      }
    } else {
      if (s >= 1) {                    // layer 1 runs one step behind layer 0
        int t = s - 1;
        v16h Ah0a = load_A_lds(lds_h0[t & 1], 0, lane);
        v16h Ah0b = load_A_lds(lds_h0[t & 1], 1, lane);
        v8f acc = cbv;
        acc = wmma_f16(Ah0a, Bih0, acc);
        acc = wmma_f16(Ah0b, Bih1, acc);
        acc = wmma_f16(Ahp0, Bhh0, acc);
        acc = wmma_f16(Ahp1, Bhh1, acc);
        v8f th;
#pragma unroll
        for (int i = 0; i < 8; ++i) th[i] = tanh_fast(acc[i]);
        store_D_lds(lds_h1[t & 1], nt, lane, th);
      }
    }
    __syncthreads();
    if (L0) {
      if (s < T_) {
        Ahp0 = load_A_lds(lds_h0[s & 1], 0, lane);
        Ahp1 = load_A_lds(lds_h0[s & 1], 1, lane);
      }
    } else {
      if (s >= 1) {
        int t = s - 1;
        Ahp0 = load_A_lds(lds_h1[t & 1], 0, lane);
        Ahp1 = load_A_lds(lds_h1[t & 1], 1, lane);
        // Persist h1(t) in A-register byte layout for the head kernel.
        half_t* rec = h1ws + ((size_t)btile * T_ + t) * 1024;
        if (w == 4)      store_A_ws(rec, 0, lane, Ahp0);
        else if (w == 5) store_A_ws(rec, 1, lane, Ahp1);
      }
    }
  }
}

// ------------------------------------------------------------ MLP head ----

__global__ __launch_bounds__(128, 4)
void head_kernel(const half_t* __restrict__ h1ws,
                 const float* __restrict__ W1, const float* __restrict__ b1,
                 const float* __restrict__ W2, const float* __restrict__ b2,
                 float* __restrict__ out) {
  __shared__ __align__(16) half_t lds_z[16 * 64];
  const int lane = threadIdx.x & 31;
  const int w    = threadIdx.x >> 5;   // 0..3

  v16h BW1a = load_B_f32(W1, w, 0, lane);
  v16h BW1b = load_B_f32(W1, w, 1, lane);
  float cb1 = b1[w * 16 + (lane & 15)];

  v16h BW2a = zero16h(), BW2b = zero16h();
  float cb2 = 0.0f;
  if (w == 0) {
    int o = lane & 15;                  // W2 has only O_=2 valid rows
    if (o < O_) {
      BW2a = load_B_f32(W2, 0, 0, lane);
      BW2b = load_B_f32(W2, 0, 1, lane);
      cb2  = b2[o];
    }
  }

  const int nTiles = BT_TILES * T_;     // 32768
  for (int tile = blockIdx.x; tile < nTiles; tile += gridDim.x) {
    int btile = tile >> 11;             // tile / T_
    int t     = tile & (T_ - 1);
    const half_t* rec = h1ws + (size_t)tile * 1024;

    int nxt = tile + gridDim.x;
    if (nxt < nTiles)
      __builtin_prefetch(h1ws + (size_t)nxt * 1024 + lane * 64, 0, 1);

    v16h A0 = load_A_ws(rec, 0, lane);  // global_load_b128 x2
    v16h A1 = load_A_ws(rec, 1, lane);
    v8f acc = splat8(cb1);
    acc = wmma_f16(A0, BW1a, acc);
    acc = wmma_f16(A1, BW1b, acc);
    v8f z;
#pragma unroll
    for (int i = 0; i < 8; ++i) z[i] = fmaxf(acc[i], 0.0f);   // ReLU
    store_D_lds(lds_z, w, lane, z);
    __syncthreads();

    if (w == 0) {
      v16h Az0 = load_A_lds(lds_z, 0, lane);
      v16h Az1 = load_A_lds(lds_z, 1, lane);
      v8f d = splat8(cb2);
      d = wmma_f16(Az0, BW2a, d);
      d = wmma_f16(Az1, BW2b, d);
      int o = lane & 15;
      if (o < O_) {
        int m0 = (lane & 16) ? 8 : 0;
        int bb = btile * 16;
#pragma unroll
        for (int i = 0; i < 8; ++i)
          out[((size_t)(bb + m0 + i) * T_ + t) * O_ + o] = d[i];
      }
    }
    __syncthreads();
  }
}

// ------------------------------------------------------------- launcher ---

extern "C" void kernel_launch(void* const* d_in, const int* in_sizes, int n_in,
                              void* d_out, int out_size, void* d_ws, size_t ws_size,
                              hipStream_t stream) {
  (void)in_sizes; (void)n_in; (void)out_size; (void)ws_size;
  const float* x    = (const float*)d_in[0];
  const float* Wih0 = (const float*)d_in[1];
  const float* Whh0 = (const float*)d_in[2];
  const float* bih0 = (const float*)d_in[3];
  const float* bhh0 = (const float*)d_in[4];
  const float* Wih1 = (const float*)d_in[5];
  const float* Whh1 = (const float*)d_in[6];
  const float* bih1 = (const float*)d_in[7];
  const float* bhh1 = (const float*)d_in[8];
  const float* W1   = (const float*)d_in[9];
  const float* b1   = (const float*)d_in[10];
  const float* W2   = (const float*)d_in[11];
  const float* b2   = (const float*)d_in[12];

  half_t* h1ws = (half_t*)d_ws;  // 16*2048 records * 2048 B = 64 MiB

  rnn2_scan_kernel<<<BT_TILES, 256, 0, stream>>>(
      x, Wih0, Whh0, bih0, bhh0, Wih1, Whh1, bih1, bhh1, h1ws);
  head_kernel<<<2048, 128, 0, stream>>>(h1ws, W1, b1, W2, b2, (float*)d_out);
}
